// SentenceAttentionModule_53601191854413
// MI455X (gfx1250) — matrix operand order repaired and checked by reading, not compiled, not run on hardware
//
#include <hip/hip_runtime.h>

// Problem constants (match reference)
#define BB   2
#define SS   2048
#define EE   1024
#define HH   16
#define HDIM 64
#define WIN  16
#define MTOT (BB * SS)   // 4096 rows

typedef __attribute__((ext_vector_type(16))) _Float16 v16h;
typedef __attribute__((ext_vector_type(8)))  float    v8f;

#define WMMA_F32_F16(a, b, c) \
  __builtin_amdgcn_wmma_f32_16x16x32_f16(false, (a), false, (b), (short)0, (c), false, false)

// ---------------------------------------------------------------------------
// Fragment loaders (CDNA5 wave32 WMMA layouts, CDNA5 ISA §7.12.2)
// ---------------------------------------------------------------------------

// 16-bit A-matrix 16x32: lane m = lane&15 holds row m; lanes 0-15 K=[k0..k0+7]
// and [k0+16..k0+23]; lanes 16-31 shifted by +8 in K.
__device__ inline v16h load_a_frag(const _Float16* __restrict__ A, int lda,
                                   int row0, int k0, int lane) {
  int m  = row0 + (lane & 15);
  int kb = k0 + ((lane >> 4) << 3);
  const _Float16* p = A + (size_t)m * lda + kb;
  v16h a;
#pragma unroll
  for (int e = 0; e < 8; ++e) a[e] = p[e];
#pragma unroll
  for (int e = 0; e < 8; ++e) a[8 + e] = p[16 + e];
  return a;
}

// B-matrix 32x16 where memory holds B transposed row-major: W[n][k]
// (torch Linear weight, K-cache [key][d] used as K^T, or V^T [d][key]).
// Lane n = lane&15 holds column n; lanes 0-15 K=[k0..k0+7] and [k0+16..k0+23],
// lanes 16-31 shifted by +8 in K (same order as the A fragment).
__device__ inline v16h load_b_fragT(const _Float16* __restrict__ W, int ldw,
                                    int n0, int k0, int lane) {
  int n  = n0 + (lane & 15);
  int kb = k0 + ((lane >> 4) << 3);
  const _Float16* p = W + (size_t)n * ldw + kb;
  v16h b;
#pragma unroll
  for (int e = 0; e < 16; ++e) b[e] = p[(e < 8) ? e : (e + 8)];
  return b;
}

// ---------------------------------------------------------------------------
// fp32 -> f16 conversion
// ---------------------------------------------------------------------------
__global__ void cvt_f32_to_f16(const float* __restrict__ src,
                               _Float16* __restrict__ dst, int n) {
  int i = blockIdx.x * blockDim.x + threadIdx.x;
  if (i < n) dst[i] = (_Float16)src[i];
}

// ---------------------------------------------------------------------------
// Kernel 1: QKV projection. X[4096,1024] x Win^T -> scatter
//   Q,K f16 [B,H,S,HD];  V f16 TRANSPOSED [B,H,HD,S] (for contiguous P.V loads)
// One wave per 16x64 strip. K-loop double-buffered; __launch_bounds__(128,1)
// widens the VGPR budget so both fragment sets stay in registers (no spills).
// ---------------------------------------------------------------------------
__global__ void __launch_bounds__(128, 1)
gemm_qkv_kernel(const _Float16* __restrict__ Xh,
                const _Float16* __restrict__ Wh,
                const float* __restrict__ bias,
                _Float16* __restrict__ Q,
                _Float16* __restrict__ K,
                _Float16* __restrict__ Vt) {
  const int NSTRIP = (3 * EE) / 64;  // 48 strips of 64 columns
  int gw   = blockIdx.x * (blockDim.x >> 5) + (threadIdx.x >> 5);
  int lane = threadIdx.x & 31;
  if (gw >= (MTOT / 16) * NSTRIP) return;
  int mt = gw / NSTRIP, st = gw % NSTRIP;
  int m0 = mt * 16, n0 = st * 64;

  v8f acc0 = {}, acc1 = {}, acc2 = {}, acc3 = {};

  // Prologue: fragments for k0 = 0
  v16h a  = load_a_frag(Xh, EE, m0, 0, lane);
  v16h b0 = load_b_fragT(Wh, EE, n0 +  0, 0, lane);
  v16h b1 = load_b_fragT(Wh, EE, n0 + 16, 0, lane);
  v16h b2 = load_b_fragT(Wh, EE, n0 + 32, 0, lane);
  v16h b3 = load_b_fragT(Wh, EE, n0 + 48, 0, lane);

#pragma unroll 2
  for (int k0 = 32; k0 < EE; k0 += 32) {
    // issue next iteration's loads first (overlap with WMMAs below)
    v16h an  = load_a_frag(Xh, EE, m0, k0, lane);
    v16h bn0 = load_b_fragT(Wh, EE, n0 +  0, k0, lane);
    v16h bn1 = load_b_fragT(Wh, EE, n0 + 16, k0, lane);
    v16h bn2 = load_b_fragT(Wh, EE, n0 + 32, k0, lane);
    v16h bn3 = load_b_fragT(Wh, EE, n0 + 48, k0, lane);

    acc0 = WMMA_F32_F16(a, b0, acc0);
    acc1 = WMMA_F32_F16(a, b1, acc1);
    acc2 = WMMA_F32_F16(a, b2, acc2);
    acc3 = WMMA_F32_F16(a, b3, acc3);

    a = an; b0 = bn0; b1 = bn1; b2 = bn2; b3 = bn3;
  }
  // Epilogue: last K chunk
  acc0 = WMMA_F32_F16(a, b0, acc0);
  acc1 = WMMA_F32_F16(a, b1, acc1);
  acc2 = WMMA_F32_F16(a, b2, acc2);
  acc3 = WMMA_F32_F16(a, b3, acc3);

  int rbase = (lane >> 4) << 3;
#pragma unroll
  for (int t = 0; t < 4; ++t) {
    v8f acc = (t == 0) ? acc0 : (t == 1) ? acc1 : (t == 2) ? acc2 : acc3;
    int col = n0 + t * 16 + (lane & 15);
    int sec = col >> 10;          // 0=Q 1=K 2=V
    int e   = col & (EE - 1);
    int h   = e >> 6;             // / HDIM
    int d   = e & (HDIM - 1);
    float bv = bias[col];
#pragma unroll
    for (int r = 0; r < 8; ++r) {
      int row = m0 + r + rbase;
      int b_  = row >> 11;          // / SS
      int s_  = row & (SS - 1);
      _Float16 val = (_Float16)(acc[r] + bv);
      if (sec == 2) {
        // V transposed: [B,H,HD,S]
        Vt[(((size_t)b_ * HH + h) * HDIM + d) * SS + s_] = val;
      } else {
        _Float16* dst = (sec == 0) ? Q : K;
        dst[(((size_t)b_ * HH + h) * SS + s_) * HDIM + d] = val;
      }
    }
  }
}

// ---------------------------------------------------------------------------
// Kernel 2: attention with inverted-band mask (blocked: i-16 <= j <= i).
// One wave handles 16 queries x HD=64, flash-softmax over 32-key chunks.
// V-fragment loads hoisted before the LDS softmax so they are in flight
// across both barriers and the exp work.
// ---------------------------------------------------------------------------
#define AW 4  // waves per block
__global__ void __launch_bounds__(AW * 32, 1)
attn_kernel(const _Float16* __restrict__ Q,
            const _Float16* __restrict__ K,
            const _Float16* __restrict__ Vt,
            _Float16* __restrict__ Ctx) {
  __shared__ float sc[AW][16][32];   // score / probability staging tile
  __shared__ float st[AW][16];       // per-row rescale factors / inv-sums

  int wid  = threadIdx.x >> 5;
  int lane = threadIdx.x & 31;
  int gw   = blockIdx.x * AW + wid;
  int qt   = gw & (SS / 16 - 1);
  int bh   = gw / (SS / 16);
  int h    = bh % HH;
  int b    = bh / HH;

  const _Float16* Qp  = Q  + ((size_t)b * HH + h) * SS * HDIM;
  const _Float16* Kp  = K  + ((size_t)b * HH + h) * SS * HDIM;
  const _Float16* Vtp = Vt + ((size_t)b * HH + h) * HDIM * SS;  // [HD][S]
  int q0 = qt * 16;

  int rbase = (lane >> 4) << 3;   // C/D fragment row offset (0 or 8)
  int ccol  = lane & 15;          // C/D fragment column

  // Q fragments for the whole loop (K-dim = HDIM = 64 -> two 32-chunks)
  v16h qa0 = load_a_frag(Qp, HDIM, q0, 0, lane);
  v16h qa1 = load_a_frag(Qp, HDIM, q0, 32, lane);

  v8f acc0 = {}, acc1 = {}, acc2 = {}, acc3 = {};
  float m_i = -3.0e30f, l_i = 0.0f;
  const float scale = 0.125f;  // 1/sqrt(64)

  for (int kt = 0; kt < SS; kt += 32) {
    // ---- scores: 16 queries x 32 keys, f32 accumulate -------------------
    v8f s0 = {}, s1 = {};
    {
      v16h kb0 = load_b_fragT(Kp, HDIM, kt, 0, lane);
      v16h kb1 = load_b_fragT(Kp, HDIM, kt, 32, lane);
      s0 = WMMA_F32_F16(qa0, kb0, s0);
      s0 = WMMA_F32_F16(qa1, kb1, s0);
      v16h kb2 = load_b_fragT(Kp, HDIM, kt + 16, 0, lane);
      v16h kb3 = load_b_fragT(Kp, HDIM, kt + 16, 32, lane);
      s1 = WMMA_F32_F16(qa0, kb2, s1);
      s1 = WMMA_F32_F16(qa1, kb3, s1);
    }

    // ---- V fragments (independent of LDS section; issue early) ----------
    v16h vb0 = load_b_fragT(Vtp, SS,  0, kt, lane);
    v16h vb1 = load_b_fragT(Vtp, SS, 16, kt, lane);
    v16h vb2 = load_b_fragT(Vtp, SS, 32, kt, lane);
    v16h vb3 = load_b_fragT(Vtp, SS, 48, kt, lane);

    __syncthreads();
    // ---- scale + mask into LDS (True = blocked band, per reference) -----
#pragma unroll
    for (int r = 0; r < 8; ++r) {
      int row = r + rbase;
      int i  = q0 + row;
      int j0 = kt + ccol;
      int j1 = kt + 16 + ccol;
      float v0 = s0[r] * scale;
      float v1 = s1[r] * scale;
      if (j0 <= i && j0 >= i - WIN) v0 = -3.0e30f;
      if (j1 <= i && j1 >= i - WIN) v1 = -3.0e30f;
      sc[wid][row][ccol]      = v0;
      sc[wid][row][16 + ccol] = v1;
    }
    __syncthreads();

    // ---- online softmax (lane L < 16 owns row L) ------------------------
    if (lane < 16) {
      float mx = m_i;
#pragma unroll
      for (int c = 0; c < 32; ++c) mx = fmaxf(mx, sc[wid][lane][c]);
      float resc = __expf(m_i - mx);
      float sum = 0.0f;
#pragma unroll
      for (int c = 0; c < 32; ++c) {
        float p = __expf(sc[wid][lane][c] - mx);
        sc[wid][lane][c] = p;
        sum += p;
      }
      l_i = l_i * resc + sum;
      m_i = mx;
      st[wid][lane] = resc;
    }
    __syncthreads();

    // ---- rescale running accumulators -----------------------------------
#pragma unroll
    for (int r = 0; r < 8; ++r) {
      float f = st[wid][r + rbase];
      acc0[r] *= f; acc1[r] *= f; acc2[r] *= f; acc3[r] *= f;
    }

    // ---- P fragment (A layout, 16x32) from LDS --------------------------
    v16h pa;
    {
      int m  = lane & 15;
      int kb = (lane >> 4) << 3;
#pragma unroll
      for (int e = 0; e < 8; ++e) pa[e]     = (_Float16)sc[wid][m][kb + e];
#pragma unroll
      for (int e = 0; e < 8; ++e) pa[8 + e] = (_Float16)sc[wid][m][kb + 16 + e];
    }

    // ---- P (16x32) x V (32x16): V^T rows are contiguous keys ------------
    acc0 = WMMA_F32_F16(pa, vb0, acc0);
    acc1 = WMMA_F32_F16(pa, vb1, acc1);
    acc2 = WMMA_F32_F16(pa, vb2, acc2);
    acc3 = WMMA_F32_F16(pa, vb3, acc3);
  }

  // ---- normalize and store context as [B,S,E] f16 (E = h*HD + d) --------
  if (lane < 16) st[wid][lane] = (l_i > 0.0f) ? (1.0f / l_i) : 0.0f;
  __syncthreads();
#pragma unroll
  for (int r = 0; r < 8; ++r) {
    float inv = st[wid][r + rbase];
    int srow  = q0 + r + rbase;
    size_t base = ((size_t)b * SS + srow) * EE + (size_t)h * HDIM;
    Ctx[base +  0 + ccol] = (_Float16)(acc0[r] * inv);
    Ctx[base + 16 + ccol] = (_Float16)(acc1[r] * inv);
    Ctx[base + 32 + ccol] = (_Float16)(acc2[r] * inv);
    Ctx[base + 48 + ccol] = (_Float16)(acc3[r] * inv);
  }
}

// ---------------------------------------------------------------------------
// Kernel 3: output projection Ctx[4096,1024] x Wout^T + bias -> fp32 out
// One wave per 16x64 strip, double-buffered K-loop (same as kernel 1).
// ---------------------------------------------------------------------------
__global__ void __launch_bounds__(128, 1)
gemm_out_kernel(const _Float16* __restrict__ Ctx,
                const _Float16* __restrict__ Wh,
                const float* __restrict__ bias,
                float* __restrict__ Out) {
  const int NSTRIP = EE / 64;  // 16 strips
  int gw   = blockIdx.x * (blockDim.x >> 5) + (threadIdx.x >> 5);
  int lane = threadIdx.x & 31;
  if (gw >= (MTOT / 16) * NSTRIP) return;
  int mt = gw / NSTRIP, st = gw % NSTRIP;
  int m0 = mt * 16, n0 = st * 64;

  v8f acc0 = {}, acc1 = {}, acc2 = {}, acc3 = {};

  v16h a  = load_a_frag(Ctx, EE, m0, 0, lane);
  v16h b0 = load_b_fragT(Wh, EE, n0 +  0, 0, lane);
  v16h b1 = load_b_fragT(Wh, EE, n0 + 16, 0, lane);
  v16h b2 = load_b_fragT(Wh, EE, n0 + 32, 0, lane);
  v16h b3 = load_b_fragT(Wh, EE, n0 + 48, 0, lane);

#pragma unroll 2
  for (int k0 = 32; k0 < EE; k0 += 32) {
    v16h an  = load_a_frag(Ctx, EE, m0, k0, lane);
    v16h bn0 = load_b_fragT(Wh, EE, n0 +  0, k0, lane);
    v16h bn1 = load_b_fragT(Wh, EE, n0 + 16, k0, lane);
    v16h bn2 = load_b_fragT(Wh, EE, n0 + 32, k0, lane);
    v16h bn3 = load_b_fragT(Wh, EE, n0 + 48, k0, lane);

    acc0 = WMMA_F32_F16(a, b0, acc0);
    acc1 = WMMA_F32_F16(a, b1, acc1);
    acc2 = WMMA_F32_F16(a, b2, acc2);
    acc3 = WMMA_F32_F16(a, b3, acc3);

    a = an; b0 = bn0; b1 = bn1; b2 = bn2; b3 = bn3;
  }
  acc0 = WMMA_F32_F16(a, b0, acc0);
  acc1 = WMMA_F32_F16(a, b1, acc1);
  acc2 = WMMA_F32_F16(a, b2, acc2);
  acc3 = WMMA_F32_F16(a, b3, acc3);

  int rbase = (lane >> 4) << 3;
#pragma unroll
  for (int t = 0; t < 4; ++t) {
    v8f acc = (t == 0) ? acc0 : (t == 1) ? acc1 : (t == 2) ? acc2 : acc3;
    int col = n0 + t * 16 + (lane & 15);
    float bv = bias[col];
#pragma unroll
    for (int r = 0; r < 8; ++r) {
      int row = m0 + r + rbase;
      Out[(size_t)row * EE + col] = acc[r] + bv;
    }
  }
}

// ---------------------------------------------------------------------------
// Host launcher
// ---------------------------------------------------------------------------
extern "C" void kernel_launch(void* const* d_in, const int* in_sizes, int n_in,
                              void* d_out, int out_size, void* d_ws, size_t ws_size,
                              hipStream_t stream) {
  const float* x     = (const float*)d_in[0];  // [B,S,E]
  /* d_in[1] = token_ids, unused by reference math */
  const float* w_in  = (const float*)d_in[2];  // [3E,E]
  const float* b_in  = (const float*)d_in[3];  // [3E]
  const float* w_out = (const float*)d_in[4];  // [E,E]
  const float* b_out = (const float*)d_in[5];  // [E]
  float* out = (float*)d_out;

  char* ws = (char*)d_ws;
  _Float16* Xh    = (_Float16*)ws; ws += (size_t)MTOT * EE * sizeof(_Float16);
  _Float16* Winh  = (_Float16*)ws; ws += (size_t)3 * EE * EE * sizeof(_Float16);
  _Float16* Wouth = (_Float16*)ws; ws += (size_t)EE * EE * sizeof(_Float16);
  _Float16* Qh    = (_Float16*)ws; ws += (size_t)MTOT * EE * sizeof(_Float16);
  _Float16* Kh    = (_Float16*)ws; ws += (size_t)MTOT * EE * sizeof(_Float16);
  _Float16* Vth   = (_Float16*)ws; ws += (size_t)MTOT * EE * sizeof(_Float16);
  _Float16* Ctxh  = (_Float16*)ws; ws += (size_t)MTOT * EE * sizeof(_Float16);

  // 1) fp32 -> f16 conversions
  {
    int n1 = MTOT * EE;           // 4,194,304
    int n2 = 3 * EE * EE;         // 3,145,728
    int n3 = EE * EE;             // 1,048,576
    cvt_f32_to_f16<<<(n1 + 255) / 256, 256, 0, stream>>>(x, Xh, n1);
    cvt_f32_to_f16<<<(n2 + 255) / 256, 256, 0, stream>>>(w_in, Winh, n2);
    cvt_f32_to_f16<<<(n3 + 255) / 256, 256, 0, stream>>>(w_out, Wouth, n3);
  }

  // 2) QKV projection: 256 M-tiles x 48 strips = 12288 waves, 4 waves/block
  {
    int waves = (MTOT / 16) * ((3 * EE) / 64);
    gemm_qkv_kernel<<<waves / 4, 128, 0, stream>>>(Xh, Winh, b_in, Qh, Kh, Vth);
  }

  // 3) attention: B*H*(S/16) = 4096 waves, 4 waves/block
  {
    int waves = BB * HH * (SS / 16);
    attn_kernel<<<waves / AW, AW * 32, 0, stream>>>(Qh, Kh, Vth, Ctxh);
  }

  // 4) output projection: 256 M-tiles x 16 strips = 4096 waves
  {
    int waves = (MTOT / 16) * (EE / 64);
    gemm_out_kernel<<<waves / 4, 128, 0, stream>>>(Ctxh, Wouth, b_out, out);
  }
}
